// BiasedAxialAttention_31224412242585
// MI455X (gfx1250) — matrix operand order repaired and checked by reading, not compiled, not run on hardware
//
#include <hip/hip_runtime.h>
#include <hip/hip_bf16.h>
#include <math.h>

// ---------------------------------------------------------------------------
// BiasedAxialAttention (row path) for gfx1250: all GEMMs on v_wmma_f32_16x16x32_bf16
// Round 2: K-split logits GEMM (8 segments -> 4608 waves instead of 576),
// partial-sum + bias reduction fused into softmax, non-temporal hints on
// write-once / read-once streams.
// ---------------------------------------------------------------------------

#define L_ 384
#define DPAIR_ 128
#define DBIAS_ 128
#define H_ 4
#define DH_ 32
#define KQK_ (L_ * DH_)   // 12288 : contraction length of the logits GEMM
#define NSEG_ 8
#define KSEG_ (KQK_ / NSEG_)  // 1536 per K-split segment

static constexpr float SCALING_ = 0.17677669529663687f;   // 1/sqrt(32)
static constexpr float RS_L_    = 0.05103103630798287f;   // 1/sqrt(384)

typedef __attribute__((ext_vector_type(16))) __bf16 v16bf;
typedef __attribute__((ext_vector_type(8)))  __bf16 v8bf;
typedef __attribute__((ext_vector_type(4)))  __bf16 v4bf;
typedef __attribute__((ext_vector_type(8)))  float  v8f;
typedef __attribute__((ext_vector_type(4)))  float  v4f;

// ---- WMMA helpers ----------------------------------------------------------

__device__ __forceinline__ v8f wmma_bf16(v16bf a, v16bf b, v8f c) {
  return __builtin_amdgcn_wmma_f32_16x16x32_bf16(
      /*neg_a=*/false, a, /*neg_b=*/false, b,
      /*c_mod=*/(short)0, c, /*reuse_a=*/false, /*reuse_b=*/false);
}

// A-matrix 16x32 bf16 fragment: lanes 0-15 hold K 0..15 of
// row (lane&15); lanes 16-31 hold K 16..31.  Two contiguous 16B loads.
__device__ __forceinline__ v16bf load_a_frag(const __bf16* base, size_t stride,
                                             int row0, int k0) {
  int lane = threadIdx.x & 31;
  int r = lane & 15;
  int hi = lane >> 4;
  const __bf16* p = base + (size_t)(row0 + r) * stride + (size_t)(k0 + hi * 16);
  v8bf lo = *reinterpret_cast<const v8bf*>(p);
  v8bf hh = *reinterpret_cast<const v8bf*>(p + 8);
  v16bf a;
#pragma unroll
  for (int i = 0; i < 8; i++) { a[i] = lo[i]; a[i + 8] = hh[i]; }
  return a;
}

// B-matrix 32x16 fragment from an N-major source S[n][k] (B[k][n] = S[n][k]):
// lane (n = lane&15, khalf = lane>>4) reads 16 contiguous K elements.
__device__ __forceinline__ v16bf load_b_frag_nmajor(const __bf16* base, size_t stride,
                                                    int col0, int k0) {
  int lane = threadIdx.x & 31;
  int n = lane & 15;
  int kh = lane >> 4;
  const __bf16* p = base + (size_t)(col0 + n) * stride + (size_t)(k0 + kh * 16);
  v8bf lo = *reinterpret_cast<const v8bf*>(p);
  v8bf hh = *reinterpret_cast<const v8bf*>(p + 8);
  v16bf b;
#pragma unroll
  for (int i = 0; i < 8; i++) { b[i] = lo[i]; b[i + 8] = hh[i]; }
  return b;
}

// Same, but source is an fp32 N-major weight matrix -> cvt to bf16 on the fly.
__device__ __forceinline__ v16bf load_b_frag_nmajor_f32(const float* base, size_t stride,
                                                        int col0, int k0) {
  int lane = threadIdx.x & 31;
  int n = lane & 15;
  int kh = lane >> 4;
  const float* p = base + (size_t)(col0 + n) * stride + (size_t)(k0 + kh * 16);
  v4f c0 = *reinterpret_cast<const v4f*>(p + 0);
  v4f c1 = *reinterpret_cast<const v4f*>(p + 4);
  v4f c2 = *reinterpret_cast<const v4f*>(p + 8);
  v4f c3 = *reinterpret_cast<const v4f*>(p + 12);
  v16bf b;
#pragma unroll
  for (int i = 0; i < 4; i++) {
    b[i + 0]  = (__bf16)c0[i];
    b[i + 4]  = (__bf16)c1[i];
    b[i + 8]  = (__bf16)c2[i];
    b[i + 12] = (__bf16)c3[i];
  }
  return b;
}

// ---- K1: LayerNorm of transposed pair -> bf16 p_ln -------------------------
// One wave per output row: p[x,y,:] = LN(pair[y,x,:]).
__global__ void baa_ln_kernel(const float* __restrict__ pair,
                              const float* __restrict__ ln_scale,
                              const float* __restrict__ ln_bias,
                              __bf16* __restrict__ pln) {
  int gw = (int)((blockIdx.x * blockDim.x + threadIdx.x) >> 5);
  int lane = threadIdx.x & 31;
  if (gw >= L_ * L_) return;
  int x = gw / L_, y = gw % L_;
  const float* src = pair + ((size_t)y * L_ + x) * DPAIR_;
  v4f v = *reinterpret_cast<const v4f*>(src + lane * 4);
  float s  = v[0] + v[1] + v[2] + v[3];
  float s2 = v[0]*v[0] + v[1]*v[1] + v[2]*v[2] + v[3]*v[3];
#pragma unroll
  for (int m = 16; m >= 1; m >>= 1) {
    s  += __shfl_xor(s,  m, 32);
    s2 += __shfl_xor(s2, m, 32);
  }
  float mu   = s * (1.0f / DPAIR_);
  float var  = s2 * (1.0f / DPAIR_) - mu * mu;
  float rinv = rsqrtf(var + 1e-5f);
  v4f sc = *reinterpret_cast<const v4f*>(ln_scale + lane * 4);
  v4f bb = *reinterpret_cast<const v4f*>(ln_bias + lane * 4);
  v4bf o;
#pragma unroll
  for (int j = 0; j < 4; j++)
    o[j] = (__bf16)((v[j] - mu) * rinv * sc[j] + bb[j]);
  *reinterpret_cast<v4bf*>(pln + (size_t)gw * DPAIR_ + lane * 4) = o;
}

// ---- K2: fused QKVG projections (WMMA, K=128) -------------------------------
// Block = 16 rows of p_ln x 512 output cols (q|k|v|g); wave w covers 64 cols.
__global__ void baa_qkvg_kernel(const __bf16* __restrict__ pln,
                                const float* __restrict__ Wq,
                                const float* __restrict__ Wk,
                                const float* __restrict__ Wv,
                                const float* __restrict__ Wg,
                                const float* __restrict__ bg,
                                __bf16* __restrict__ qT,
                                __bf16* __restrict__ kT,
                                __bf16* __restrict__ vT,
                                __bf16* __restrict__ g) {
  int w = threadIdx.x >> 5;               // 0..7
  int row0 = blockIdx.x * 16;             // row in flattened [L*L] p space
  int proj = w >> 1;                      // 0=q 1=k 2=v 3=g
  int colbase = (w & 1) * 64;             // within 128 cols of that projection
  const float* W = (proj == 0) ? Wq : (proj == 1) ? Wk : (proj == 2) ? Wv : Wg;

  v8f acc[4] = {v8f{}, v8f{}, v8f{}, v8f{}};
#pragma unroll
  for (int kk = 0; kk < DPAIR_; kk += 32) {
    v16bf a = load_a_frag(pln, DPAIR_, row0, kk);
#pragma unroll
    for (int t = 0; t < 4; t++) {
      v16bf b = load_b_frag_nmajor_f32(W, DPAIR_, colbase + t * 16, kk);
      acc[t] = wmma_bf16(a, b, acc[t]);
    }
  }

  int lane = threadIdx.x & 31;
  int n = lane & 15, hi = lane >> 4;
  int nrow  = row0 / L_;   // outer row axis (tiles never cross it: 384 % 16 == 0)
  int mbase = row0 % L_;
#pragma unroll
  for (int t = 0; t < 4; t++) {
    int ocol = colbase + t * 16 + n;  // 0..127 inside this projection
    int hh = ocol >> 5, dd = ocol & 31;
    if (proj == 2) {
      // v: pack 8 consecutive j into one 16B store of vT[(kr,h,d)][j]
      v8bf pk;
#pragma unroll
      for (int r = 0; r < 8; r++) pk[r] = (__bf16)acc[t][r];
      size_t idx = (((size_t)nrow * H_ + hh) * DH_ + dd) * L_ + mbase + hi * 8;
      *reinterpret_cast<v8bf*>(vT + idx) = pk;
    } else {
#pragma unroll
      for (int r = 0; r < 8; r++) {
        int mm = mbase + r + hi * 8;
        float val = acc[t][r];
        if (proj == 0) {
          qT[(((size_t)hh * L_ + mm) * L_ + nrow) * DH_ + dd] = (__bf16)(val * SCALING_);
        } else if (proj == 1) {
          kT[(((size_t)hh * L_ + mm) * L_ + nrow) * DH_ + dd] = (__bf16)(val * RS_L_);
        } else {
          float gv = 1.0f / (1.0f + __expf(-(val + bg[ocol])));
          g[((size_t)nrow * L_ + mm) * DPAIR_ + ocol] = (__bf16)gv;
        }
      }
    }
  }
}

// ---- K3: pair-bias projection blog[h][i][j] --------------------------------
__global__ void baa_biasproj_kernel(const float* __restrict__ bias,
                                    const float* __restrict__ Wb,
                                    float* __restrict__ blog) {
  int gw = (int)((blockIdx.x * blockDim.x + threadIdx.x) >> 5);  // i*L + j
  int lane = threadIdx.x & 31;
  if (gw >= L_ * L_) return;
  v4f x = *reinterpret_cast<const v4f*>(bias + (size_t)gw * DBIAS_ + lane * 4);
#pragma unroll
  for (int h = 0; h < H_; h++) {
    v4f wv = *reinterpret_cast<const v4f*>(Wb + h * DBIAS_ + lane * 4);
    float p = x[0]*wv[0] + x[1]*wv[1] + x[2]*wv[2] + x[3]*wv[3];
#pragma unroll
    for (int m = 16; m >= 1; m >>= 1) p += __shfl_xor(p, m, 32);
    if (lane == 0) blog[(size_t)h * L_ * L_ + gw] = p;
  }
}

// ---- K4a: K-split logits GEMM (M=i, N=j, K-segment of 1536) ----------------
// Wave = one 16(i) x 64(j) tile of one head for one of 8 K-segments.
// 4 x 24 x 6 x 8 = 4608 waves: fills the chip (was 576 without the split).
// Partials go to attnP[seg] with non-temporal stores (write-once stream).
__global__ void baa_logits_kernel(const __bf16* __restrict__ qT,
                                  const __bf16* __restrict__ kT,
                                  float* __restrict__ attnP) {
  int gw = (int)((blockIdx.x * blockDim.x + threadIdx.x) >> 5);
  int seg = gw % NSEG_; int tmp = gw / NSEG_;
  int jg = tmp % 6; tmp /= 6;
  int it = tmp % 24; int h = tmp / 24;
  if (h >= H_) return;
  const __bf16* qh = qT + (size_t)h * L_ * KQK_;
  const __bf16* kh = kT + (size_t)h * L_ * KQK_;
  int i0 = it * 16, j0 = jg * 64;
  int kbeg = seg * KSEG_, kend = kbeg + KSEG_;

  v8f acc[4] = {v8f{}, v8f{}, v8f{}, v8f{}};
  for (int kk = kbeg; kk < kend; kk += 32) {
    if (kk + 256 < kend) {
      int lane = threadIdx.x & 31;
      __builtin_prefetch(qh + (size_t)(i0 + (lane & 15)) * KQK_ + kk + 256, 0, 1);
    }
    v16bf a = load_a_frag(qh, KQK_, i0, kk);
#pragma unroll
    for (int t = 0; t < 4; t++) {
      v16bf b = load_b_frag_nmajor(kh, KQK_, j0 + t * 16, kk);
      acc[t] = wmma_bf16(a, b, acc[t]);
    }
  }

  float* dst = attnP + (size_t)seg * H_ * L_ * L_ + (size_t)h * L_ * L_;
  int lane = threadIdx.x & 31;
  int n = lane & 15, hi = lane >> 4;
#pragma unroll
  for (int t = 0; t < 4; t++) {
#pragma unroll
    for (int r = 0; r < 8; r++) {
      int i = i0 + r + hi * 8;
      int j = j0 + t * 16 + n;
      __builtin_nontemporal_store(acc[t][r], dst + (size_t)i * L_ + j);
    }
  }
}

// ---- K4b: partial reduce + bias + softmax over j, write bf16 ---------------
__global__ void baa_softmax_kernel(const float* __restrict__ attnP,
                                   const float* __restrict__ blog,
                                   __bf16* __restrict__ attnS) {
  int gw = (int)((blockIdx.x * blockDim.x + threadIdx.x) >> 5);  // h*L + i
  int lane = threadIdx.x & 31;
  if (gw >= H_ * L_) return;
  const size_t rowoff = (size_t)gw * L_;
  const size_t segstride = (size_t)H_ * L_ * L_;

  float vals[12];
  float mx = -1e30f;
#pragma unroll
  for (int t = 0; t < 12; t++) {
    int j = lane + t * 32;
    float a = blog[rowoff + j];
#pragma unroll
    for (int s = 0; s < NSEG_; s++)
      a += __builtin_nontemporal_load(attnP + (size_t)s * segstride + rowoff + j);
    vals[t] = a;
    mx = fmaxf(mx, a);
  }
#pragma unroll
  for (int m = 16; m >= 1; m >>= 1) mx = fmaxf(mx, __shfl_xor(mx, m, 32));
  float s = 0.0f;
#pragma unroll
  for (int t = 0; t < 12; t++) { vals[t] = __expf(vals[t] - mx); s += vals[t]; }
#pragma unroll
  for (int m = 16; m >= 1; m >>= 1) s += __shfl_xor(s, m, 32);
  float inv = 1.0f / s;
  __bf16* orow = attnS + rowoff;
#pragma unroll
  for (int t = 0; t < 12; t++) orow[lane + t * 32] = (__bf16)(vals[t] * inv);
}

// ---- K5: attn @ V, gate multiply (per (kr,h): M=i, N=d=32, K=j=384) --------
__global__ void baa_attnv_kernel(const __bf16* __restrict__ attnS,
                                 const __bf16* __restrict__ vT,
                                 const __bf16* __restrict__ g,
                                 __bf16* __restrict__ outPre) {
  int gw = (int)((blockIdx.x * blockDim.x + threadIdx.x) >> 5);
  int it = gw % 24; int tmp = gw / 24;
  int h = tmp % H_; int kr = tmp / H_;
  if (kr >= L_) return;
  int i0 = it * 16;
  const __bf16* ah = attnS + (size_t)h * L_ * L_;
  const __bf16* vh = vT + ((size_t)kr * H_ + h) * DH_ * L_;  // rows=d, stride L

  v8f acc[2] = {v8f{}, v8f{}};
  for (int kk = 0; kk < L_; kk += 32) {
    v16bf a = load_a_frag(ah, L_, i0, kk);
    acc[0] = wmma_bf16(a, load_b_frag_nmajor(vh, L_, 0, kk), acc[0]);
    acc[1] = wmma_bf16(a, load_b_frag_nmajor(vh, L_, 16, kk), acc[1]);
  }

  int lane = threadIdx.x & 31;
  int n = lane & 15, hi = lane >> 4;
#pragma unroll
  for (int t = 0; t < 2; t++) {
#pragma unroll
    for (int r = 0; r < 8; r++) {
      int i = i0 + r + hi * 8;
      int o = h * DH_ + t * 16 + n;
      size_t idx = ((size_t)i * L_ + kr) * DPAIR_ + o;
      float gv = (float)g[idx];
      outPre[idx] = (__bf16)(acc[t][r] * gv);
    }
  }
}

// ---- K6: output projection + bias + final transpose ------------------------
__global__ void baa_outproj_kernel(const __bf16* __restrict__ outPre,
                                   const float* __restrict__ Wo,
                                   const float* __restrict__ bo,
                                   float* __restrict__ out) {
  int w = threadIdx.x >> 5;       // 0..7 -> 16-wide col tile
  int row0 = blockIdx.x * 16;     // row in flattened [i*L + kr]
  int col0 = w * 16;

  v8f acc = v8f{};
#pragma unroll
  for (int kk = 0; kk < DPAIR_; kk += 32) {
    v16bf a = load_a_frag(outPre, DPAIR_, row0, kk);
    v16bf b = load_b_frag_nmajor_f32(Wo, DPAIR_, col0, kk);
    acc = wmma_bf16(a, b, acc);
  }

  int lane = threadIdx.x & 31;
  int n = lane & 15, hi = lane >> 4;
  int o = col0 + n;
  float bv = bo[o];
  int i = row0 / L_;
  int krbase = row0 % L_;
#pragma unroll
  for (int r = 0; r < 8; r++) {
    int kr = krbase + r + hi * 8;
    // reference returns transpose(0,2,1,3): d_out[kr][i][o]
    __builtin_nontemporal_store(acc[r] + bv, out + ((size_t)kr * L_ + i) * DPAIR_ + o);
  }
}

// ---------------------------------------------------------------------------

extern "C" void kernel_launch(void* const* d_in, const int* in_sizes, int n_in,
                              void* d_out, int out_size, void* d_ws, size_t ws_size,
                              hipStream_t stream) {
  (void)in_sizes; (void)n_in; (void)out_size; (void)ws_size;

  const float* pair     = (const float*)d_in[0];
  const float* bias     = (const float*)d_in[1];
  const float* ln_scale = (const float*)d_in[2];
  const float* ln_bias  = (const float*)d_in[3];
  const float* Wq       = (const float*)d_in[4];
  const float* Wk       = (const float*)d_in[5];
  const float* Wv       = (const float*)d_in[6];
  const float* Wb       = (const float*)d_in[7];
  const float* Wg       = (const float*)d_in[8];
  const float* bg       = (const float*)d_in[9];
  const float* Wo       = (const float*)d_in[10];
  const float* bo       = (const float*)d_in[11];
  float* out            = (float*)d_out;

  // Workspace layout (bytes); NP = L*L*128 bf16 elements = 37,748,736 B.
  const size_t NP  = (size_t)L_ * L_ * DPAIR_ * sizeof(__bf16);
  const size_t NAT = (size_t)H_ * L_ * L_ * sizeof(float);   // 2,359,296 B
  char* ws = (char*)d_ws;
  __bf16* pln    = (__bf16*)(ws + 0 * NP);          // reused as outPre after K2
  __bf16* qT     = (__bf16*)(ws + 1 * NP);
  __bf16* kT     = (__bf16*)(ws + 2 * NP);
  __bf16* vT     = (__bf16*)(ws + 3 * NP);
  __bf16* g      = (__bf16*)(ws + 4 * NP);
  float*  blog   = (float*)(ws + 5 * NP);
  float*  attnP  = (float*)(ws + 5 * NP + NAT);               // NSEG_ * 2.36 MB
  __bf16* attnS  = (__bf16*)(ws + 5 * NP + NAT + NSEG_ * NAT);
  __bf16* outPre = pln;  // p_ln dead after K2

  const int T = 256;  // 8 wave32s per block

  // K1: LayerNorm (one wave per row) -> 147456 waves
  baa_ln_kernel<<<(L_ * L_) / 8, T, 0, stream>>>(pair, ln_scale, ln_bias, pln);

  // K3: bias projection (independent of K1/K2; stream order is fine)
  baa_biasproj_kernel<<<(L_ * L_) / 8, T, 0, stream>>>(bias, Wb, blog);

  // K2: fused QKVG projection; block = 16 rows x 512 cols
  baa_qkvg_kernel<<<(L_ * L_) / 16, T, 0, stream>>>(pln, Wq, Wk, Wv, Wg, bg,
                                                    qT, kT, vT, g);

  // K4a: K-split logits; 4 h x 24 i-tiles x 6 j-groups x 8 segs = 4608 waves
  baa_logits_kernel<<<(H_ * 24 * 6 * NSEG_) / 8, T, 0, stream>>>(qT, kT, attnP);

  // K4b: reduce partials + bias + softmax; one wave per (h,i) row
  baa_softmax_kernel<<<(H_ * L_) / 8, T, 0, stream>>>(attnP, blog, attnS);

  // K5: attn @ V + gate; 384 kr x 4 h x 24 i-tiles = 36864 waves
  baa_attnv_kernel<<<(L_ * H_ * 24) / 8, T, 0, stream>>>(attnS, vT, g, outPre);

  // K6: output projection + transpose; block = 16 rows x 128 cols
  baa_outproj_kernel<<<(L_ * L_) / 16, T, 0, stream>>>(outPre, Wo, bo, out);
}
